// trajectory2seq_attn_30468497997931
// MI455X (gfx1250) — matrix operand order repaired and checked by reading, not compiled, not run on hardware
//
#include <hip/hip_runtime.h>
#include <hip/hip_bf16.h>
#include <cstdint>
#include <cstddef>

#define B_   64
#define TIN_ 256
#define F_   64
#define H_   1024
#define TGT_ 64
#define V_   32000

typedef __attribute__((ext_vector_type(16))) __bf16 v16bf;
typedef __attribute__((ext_vector_type(8)))  float  v8f;
typedef __attribute__((ext_vector_type(4)))  int    int4v;

union FragAB { uint4 u[2]; v16bf v; };
union U8bf  { uint4 u; __bf16 h[8]; };

#if __has_builtin(__builtin_amdgcn_global_load_async_to_lds_b128)
#define HAVE_ASYNC_LDS 1
#else
#define HAVE_ASYNC_LDS 0
#endif

__device__ __forceinline__ void wait_asynccnt0() {
#if __has_builtin(__builtin_amdgcn_s_wait_asynccnt)
  __builtin_amdgcn_s_wait_asynccnt(0);
#else
  asm volatile("s_wait_asynccnt 0x0" ::: "memory");
#endif
}

// Async copy of one 16-byte chunk global -> LDS (ASYNCcnt-tracked).
__device__ __forceinline__ void async_copy16(const __bf16* gp, __bf16* lp) {
#if HAVE_ASYNC_LDS
  __builtin_amdgcn_global_load_async_to_lds_b128(
      (__attribute__((address_space(1))) int4v*)gp,
      (__attribute__((address_space(3))) int4v*)lp, 0, 0);
#else
  *reinterpret_cast<uint4*>(lp) = *reinterpret_cast<const uint4*>(gp);
#endif
}

// A-fragment (16x32 bf16, M x K) from global: lane L holds row m = L%16,
// K chunks [k0 + 8*(L>>4), +8) and [k0 + 16 + 8*(L>>4), +8)  (ISA 7.12.2)
__device__ __forceinline__ v16bf load_a_frag(const __bf16* __restrict__ X,
                                             int lda, int mbase, int k0, int lane) {
  int m = mbase + (lane & 15);
  int khalf = lane >> 4;
  const __bf16* p = X + (size_t)m * lda + k0 + khalf * 8;
  FragAB f;
  f.u[0] = *reinterpret_cast<const uint4*>(p);
  f.u[1] = *reinterpret_cast<const uint4*>(p + 16);
  return f.v;
}

// Same A-fragment but from an LDS panel (16 rows x K, row stride K)
__device__ __forceinline__ v16bf load_a_frag_lds(const __bf16* Ap,
                                                 int K, int k0, int lane) {
  int m = lane & 15;
  int khalf = lane >> 4;
  const __bf16* p = Ap + m * K + k0 + khalf * 8;
  FragAB f;
  f.u[0] = *reinterpret_cast<const uint4*>(p);
  f.u[1] = *reinterpret_cast<const uint4*>(p + 16);
  return f.v;
}

// B-fragment (32x16 bf16, K x N) with B = W^T, W stored [N,K] row-major:
// lane L holds column n = L%16 -> row of W, K chunks 8*(L>>4) and 16+8*(L>>4).
__device__ __forceinline__ v16bf load_b_frag(const __bf16* __restrict__ W,
                                             int K, int nbase, int k0, int lane) {
  int n = nbase + (lane & 15);
  int kb = k0 + ((lane >> 4) << 3);
  const __bf16* p = W + (size_t)n * K + kb;
  FragAB f;
  f.u[0] = *reinterpret_cast<const uint4*>(p);
  f.u[1] = *reinterpret_cast<const uint4*>(p + 16);
  return f.v;
}

// ---------------- fused GRU cell ----------------
// grid (B/16, H/16), 192 threads = 6 waves: wave = gate(0..2) + 3*src(gi/gh).
// Wave-dependent operands are scalarized via readfirstlane so the K-loop is
// SALU-driven. Two accumulator chains break the WMMA RAW dependency.
__global__ __launch_bounds__(192) void gru_cell_kernel(
    const __bf16* __restrict__ X, int lda, int Kin,
    const __bf16* __restrict__ Hprev_b, const float* __restrict__ Hprev_f,
    const __bf16* __restrict__ Wih, const __bf16* __restrict__ Whh,
    const float* __restrict__ bih, const float* __restrict__ bhh,
    float* __restrict__ Hnew_f, __bf16* __restrict__ Hnew_b,
    __bf16* __restrict__ extra_b, long extra_stride)
{
  __shared__ float gbuf[2][3][16][16];
  const int tid = threadIdx.x, lane = tid & 31;
  const int wid = __builtin_amdgcn_readfirstlane(tid >> 5);  // scalar per-wave
  const int g = wid % 3;            // 0=r 1=z 2=n
  const int s = wid / 3;            // 0: gi (X*Wih), 1: gh (h*Whh)
  const int mbase = blockIdx.x * 16;
  const int ntile = blockIdx.y * 16;
  const int nbase = g * H_ + ntile;

  const __bf16* A  = (s == 0) ? X : Hprev_b;
  const int K      = (s == 0) ? Kin : H_;
  const int Alda   = (s == 0) ? lda : H_;
  const __bf16* W  = (s == 0) ? Wih : Whh;
  const float* bia = (s == 0) ? bih : bhh;

  v8f acc0 = {0.f,0.f,0.f,0.f,0.f,0.f,0.f,0.f};
  v8f acc1 = {0.f,0.f,0.f,0.f,0.f,0.f,0.f,0.f};
  const __bf16* wrow = W + (size_t)(nbase + (lane & 15)) * K;
  for (int k0 = 0; k0 < K; k0 += 64) {
    __builtin_prefetch(wrow + k0 + 64, 0, 3);   // speculative, OOB-safe
    v16bf a0 = load_a_frag(A, Alda, mbase, k0, lane);
    v16bf b0 = load_b_frag(W, K, nbase, k0, lane);
    acc0 = __builtin_amdgcn_wmma_f32_16x16x32_bf16(false, a0, false, b0,
                                                   (short)0, acc0, false, false);
    v16bf a1 = load_a_frag(A, Alda, mbase, k0 + 32, lane);
    v16bf b1 = load_b_frag(W, K, nbase, k0 + 32, lane);
    acc1 = __builtin_amdgcn_wmma_f32_16x16x32_bf16(false, a1, false, b1,
                                                   (short)0, acc1, false, false);
  }
  const float bv = bia[nbase + (lane & 15)];
  const int rowoff = (lane >> 4) * 8, col = lane & 15;
  #pragma unroll
  for (int j = 0; j < 8; ++j)
    gbuf[s][g][rowoff + j][col] = acc0[j] + acc1[j] + bv;
  __syncthreads();

  for (int idx = tid; idx < 256; idx += 192) {
    int m = idx >> 4, c = idx & 15;
    float r  = 1.f / (1.f + __expf(-(gbuf[0][0][m][c] + gbuf[1][0][m][c])));
    float z  = 1.f / (1.f + __expf(-(gbuf[0][1][m][c] + gbuf[1][1][m][c])));
    float nn = tanhf(gbuf[0][2][m][c] + r * gbuf[1][2][m][c]);
    int gm = mbase + m, gc = ntile + c;
    float hp = Hprev_f[(size_t)gm * H_ + gc];
    float hn = (1.f - z) * nn + z * hp;
    Hnew_f[(size_t)gm * H_ + gc] = hn;
    __bf16 hb = (__bf16)hn;
    Hnew_b[(size_t)gm * H_ + gc] = hb;
    if (extra_b) extra_b[(size_t)gm * extra_stride + gc] = hb;
  }
}

// ---------------- generic WMMA GEMM: out = A[M,K](bf16) * W[N,K]^T + bias ----
// grid (M/16, N/64), 128 threads = 4 waves, one 16x16 tile per wave.
// The shared 16xK A-panel is staged into LDS once via async load-to-LDS
// (ASYNCcnt path), then all 4 waves feed WMMA A-fragments from LDS.
// Dynamic LDS size = 16*K*2 bytes.
__global__ __launch_bounds__(128) void gemm_bias_kernel(
    const __bf16* __restrict__ A, int lda,
    const __bf16* __restrict__ W, int K,
    const float* __restrict__ bias,
    float* __restrict__ outf, long out_rstride, long out_coff,
    __bf16* __restrict__ outb, long outb_rstride)
{
  extern __shared__ __bf16 Apanel[];           // [16][K]
  const int tid = threadIdx.x, lane = tid & 31;
  const int wid = __builtin_amdgcn_readfirstlane(tid >> 5);
  const int mbase = blockIdx.x * 16;
  const int nbase = (blockIdx.y * 4 + wid) * 16;

  // ---- stage A panel (16 rows x K) into LDS ----
  {
    const int chunksPerRow = K >> 3;           // 16-byte chunks per row
    const int totalChunks = 16 * chunksPerRow; // multiple of 128 for K%64==0
    for (int c = tid; c < totalChunks; c += 128) {
      int m = c / chunksPerRow;
      int j = c - m * chunksPerRow;
      async_copy16(A + (size_t)(mbase + m) * lda + j * 8,
                   &Apanel[m * K + j * 8]);
    }
#if HAVE_ASYNC_LDS
    wait_asynccnt0();
#endif
    __syncthreads();
  }

  v8f acc0 = {0.f,0.f,0.f,0.f,0.f,0.f,0.f,0.f};
  v8f acc1 = {0.f,0.f,0.f,0.f,0.f,0.f,0.f,0.f};
  const __bf16* wrow = W + (size_t)(nbase + (lane & 15)) * K;
  for (int k0 = 0; k0 < K; k0 += 64) {
    __builtin_prefetch(wrow + k0 + 64, 0, 3);
    v16bf a0 = load_a_frag_lds(Apanel, K, k0, lane);
    v16bf b0 = load_b_frag(W, K, nbase, k0, lane);
    acc0 = __builtin_amdgcn_wmma_f32_16x16x32_bf16(false, a0, false, b0,
                                                   (short)0, acc0, false, false);
    v16bf a1 = load_a_frag_lds(Apanel, K, k0 + 32, lane);
    v16bf b1 = load_b_frag(W, K, nbase, k0 + 32, lane);
    acc1 = __builtin_amdgcn_wmma_f32_16x16x32_bf16(false, a1, false, b1,
                                                   (short)0, acc1, false, false);
  }
  const float bv = bias ? bias[nbase + (lane & 15)] : 0.f;
  const int rowoff = (lane >> 4) * 8, col = lane & 15;
  #pragma unroll
  for (int j = 0; j < 8; ++j) {
    int m = mbase + rowoff + j;
    int n = nbase + col;
    float v = acc0[j] + acc1[j] + bv;
    if (outf) outf[(size_t)m * out_rstride + out_coff + n] = v;
    if (outb) outb[(size_t)m * outb_rstride + n] = (__bf16)v;
  }
}

// ---------------- attention: scores, softmax, weighted sum ----------------
__global__ __launch_bounds__(256) void attn_score_kernel(
    const float* __restrict__ q, const __bf16* __restrict__ enc,
    float* __restrict__ scores)
{
  __shared__ float qs[H_];
  int b = blockIdx.x, t = threadIdx.x;
  for (int i = t; i < H_; i += 256) qs[i] = q[(size_t)b * H_ + i];
  __syncthreads();
  const uint4* e4 = reinterpret_cast<const uint4*>(
      enc + ((size_t)b * TIN_ + t) * H_);
  float acc = 0.f;
  #pragma unroll 4
  for (int h8 = 0; h8 < H_ / 8; ++h8) {
    U8bf w; w.u = e4[h8];
    #pragma unroll
    for (int j = 0; j < 8; ++j) acc += qs[h8 * 8 + j] * (float)w.h[j];
  }
  scores[(size_t)b * TIN_ + t] = acc;
}

__global__ __launch_bounds__(256) void softmax_attn_kernel(
    const float* __restrict__ scores, float* __restrict__ aw,
    float* __restrict__ attn_out, int step)
{
  __shared__ float red[256];
  int b = blockIdx.x, t = threadIdx.x;
  float v = scores[(size_t)b * TIN_ + t];
  red[t] = v; __syncthreads();
  for (int s = 128; s > 0; s >>= 1) {
    if (t < s) red[t] = fmaxf(red[t], red[t + s]);
    __syncthreads();
  }
  float mx = red[0]; __syncthreads();
  float e = __expf(v - mx);
  red[t] = e; __syncthreads();
  for (int s = 128; s > 0; s >>= 1) {
    if (t < s) red[t] += red[t + s];
    __syncthreads();
  }
  float p = e / red[0];
  aw[(size_t)b * TIN_ + t] = p;
  attn_out[(size_t)b * TIN_ * TGT_ + (size_t)t * TGT_ + step] = p;
}

__global__ __launch_bounds__(256) void att_sum_kernel(
    const float* __restrict__ aw, const __bf16* __restrict__ enc,
    __bf16* __restrict__ comb)  // writes att into comb[:, 0:H]
{
  int b = blockIdx.x, tid = threadIdx.x;
  float acc[4] = {0.f, 0.f, 0.f, 0.f};
  for (int t = 0; t < TIN_; ++t) {
    float a = aw[(size_t)b * TIN_ + t];
    const __bf16* e = enc + ((size_t)b * TIN_ + t) * H_;
    #pragma unroll
    for (int i = 0; i < 4; ++i) acc[i] += a * (float)e[tid + 256 * i];
  }
  #pragma unroll
  for (int i = 0; i < 4; ++i)
    comb[(size_t)b * (2 * H_) + tid + 256 * i] = (__bf16)acc[i];
}

// ---------------- misc small kernels ----------------
__global__ __launch_bounds__(256) void embed_kernel(
    const int* __restrict__ tok, const float* __restrict__ embed,
    __bf16* __restrict__ xdec)
{
  int b = blockIdx.x, tid = threadIdx.x;
  int tk = tok[b];
  #pragma unroll
  for (int i = 0; i < 4; ++i) {
    int h = tid + 256 * i;
    xdec[(size_t)b * H_ + h] = (__bf16)embed[(size_t)tk * H_ + h];
  }
}

__global__ __launch_bounds__(256) void argmax_kernel(
    const float* __restrict__ logits, long rstride, int* __restrict__ tok)
{
  __shared__ float bv[256];
  __shared__ int   bi[256];
  int b = blockIdx.x, t = threadIdx.x;
  const float* row = logits + (size_t)b * rstride;
  float best = -3.402823466e+38f; int bidx = 0;
  for (int v = t; v < V_; v += 256) {
    float x = row[v];
    if (x > best) { best = x; bidx = v; }
  }
  bv[t] = best; bi[t] = bidx; __syncthreads();
  for (int s = 128; s > 0; s >>= 1) {
    if (t < s) {
      if (bv[t + s] > bv[t] || (bv[t + s] == bv[t] && bi[t + s] < bi[t])) {
        bv[t] = bv[t + s]; bi[t] = bi[t + s];
      }
    }
    __syncthreads();
  }
  if (t == 0) tok[b] = bi[0];
}

__global__ void cvt_bf16_kernel(const float* __restrict__ src,
                                __bf16* __restrict__ dst, size_t n) {
  size_t i = (size_t)blockIdx.x * 256 + threadIdx.x;
  if (i < n) dst[i] = (__bf16)src[i];
}

__global__ void zero_u32_kernel(unsigned* __restrict__ p, size_t n) {
  size_t i = (size_t)blockIdx.x * 256 + threadIdx.x;
  if (i < n) p[i] = 0u;
}

__global__ void copy_hidden_kernel(const float* __restrict__ h0,
                                   const float* __restrict__ h1,
                                   float* __restrict__ out) {
  int i = blockIdx.x * 256 + threadIdx.x;
  if (i < B_ * H_) { out[i] = h0[i]; out[B_ * H_ + i] = h1[i]; }
}

// ---------------- host orchestration ----------------
extern "C" void kernel_launch(void* const* d_in, const int* in_sizes, int n_in,
                              void* d_out, int out_size, void* d_ws, size_t ws_size,
                              hipStream_t stream) {
  (void)in_sizes; (void)n_in; (void)out_size; (void)ws_size;
  const float* x       = (const float*)d_in[0];
  const float* embed   = (const float*)d_in[1];
  const float* Wih[4]  = {(const float*)d_in[2], (const float*)d_in[6],
                          (const float*)d_in[10], (const float*)d_in[14]};
  const float* Whh[4]  = {(const float*)d_in[3], (const float*)d_in[7],
                          (const float*)d_in[11], (const float*)d_in[15]};
  const float* bih[4]  = {(const float*)d_in[4], (const float*)d_in[8],
                          (const float*)d_in[12], (const float*)d_in[16]};
  const float* bhh[4]  = {(const float*)d_in[5], (const float*)d_in[9],
                          (const float*)d_in[13], (const float*)d_in[17]};
  const float* Wq  = (const float*)d_in[18]; const float* bq  = (const float*)d_in[19];
  const float* Wc  = (const float*)d_in[20]; const float* bc  = (const float*)d_in[21];
  const float* Wfc = (const float*)d_in[22]; const float* bfc = (const float*)d_in[23];
  float* dout = (float*)d_out;

  // workspace layout
  char* ws = (char*)d_ws;
  size_t off = 0;
  auto alloc = [&](size_t bytes) -> void* {
    void* p = ws + off; off = (off + bytes + 255) & ~(size_t)255; return p;
  };
  const size_t Kih[4] = {F_, H_, H_, H_};
  __bf16* Wihb[4]; __bf16* Whhb[4];
  for (int l = 0; l < 4; ++l) {
    Wihb[l] = (__bf16*)alloc((size_t)3 * H_ * Kih[l] * 2);
    Whhb[l] = (__bf16*)alloc((size_t)3 * H_ * H_ * 2);
  }
  __bf16* Wqb  = (__bf16*)alloc((size_t)H_ * H_ * 2);
  __bf16* Wcb  = (__bf16*)alloc((size_t)H_ * 2 * H_ * 2);
  __bf16* Wfcb = (__bf16*)alloc((size_t)V_ * H_ * 2);
  __bf16* xb   = (__bf16*)alloc((size_t)B_ * TIN_ * F_ * 2);
  __bf16* encb = (__bf16*)alloc((size_t)B_ * TIN_ * H_ * 2);
  float*  h0f[2]; float* h1f[2]; __bf16* h0b[2]; __bf16* h1b[2];
  for (int i = 0; i < 2; ++i) {
    h0f[i] = (float*)alloc((size_t)B_ * H_ * 4);
    h1f[i] = (float*)alloc((size_t)B_ * H_ * 4);
    h0b[i] = (__bf16*)alloc((size_t)B_ * H_ * 2);
    h1b[i] = (__bf16*)alloc((size_t)B_ * H_ * 2);
  }
  __bf16* xdec  = (__bf16*)alloc((size_t)B_ * H_ * 2);
  float*  qf    = (float*)alloc((size_t)B_ * H_ * 4);
  float*  scor  = (float*)alloc((size_t)B_ * TIN_ * 4);
  float*  aw    = (float*)alloc((size_t)B_ * TIN_ * 4);
  __bf16* combb = (__bf16*)alloc((size_t)B_ * 2 * H_ * 2);  // [att | h1]
  __bf16* cvecb = (__bf16*)alloc((size_t)B_ * H_ * 2);
  int*    tok   = (int*)alloc((size_t)B_ * 4);

  auto cvt = [&](const float* s, __bf16* d, size_t n) {
    cvt_bf16_kernel<<<dim3((unsigned)((n + 255) / 256)), 256, 0, stream>>>(s, d, n);
  };
  for (int l = 0; l < 4; ++l) {
    cvt(Wih[l], Wihb[l], (size_t)3 * H_ * Kih[l]);
    cvt(Whh[l], Whhb[l], (size_t)3 * H_ * H_);
  }
  cvt(Wq, Wqb, (size_t)H_ * H_);
  cvt(Wc, Wcb, (size_t)H_ * 2 * H_);
  cvt(Wfc, Wfcb, (size_t)V_ * H_);
  cvt(x, xb, (size_t)B_ * TIN_ * F_);

  auto zero = [&](void* p, size_t words) {
    zero_u32_kernel<<<dim3((unsigned)((words + 255) / 256)), 256, 0, stream>>>(
        (unsigned*)p, words);
  };
  zero(h0f[0], (size_t)B_ * H_); zero(h1f[0], (size_t)B_ * H_);
  zero(h0b[0], (size_t)B_ * H_ / 2); zero(h1b[0], (size_t)B_ * H_ / 2);
  zero(tok, B_);

  const dim3 gruGrid(B_ / 16, H_ / 16);
  const size_t panelH  = (size_t)16 * H_ * 2;      // 32 KB
  const size_t panel2H = (size_t)16 * 2 * H_ * 2;  // 64 KB

  // ---------------- encoder: 256 steps, 2 layers ----------------
  int pp = 0;
  for (int t = 0; t < TIN_; ++t) {
    int cur = 1 - pp;
    gru_cell_kernel<<<gruGrid, 192, 0, stream>>>(
        xb + (size_t)t * F_, TIN_ * F_, F_,
        h0b[pp], h0f[pp], Wihb[0], Whhb[0], bih[0], bhh[0],
        h0f[cur], h0b[cur], (__bf16*)nullptr, 0L);
    gru_cell_kernel<<<gruGrid, 192, 0, stream>>>(
        h0b[cur], H_, H_,
        h1b[pp], h1f[pp], Wihb[1], Whhb[1], bih[1], bhh[1],
        h1f[cur], h1b[cur], encb + (size_t)t * H_, (long)TIN_ * H_);
    pp = cur;
  }

  // ---------------- decoder: 64 autoregressive steps ----------------
  float* attn_out = dout + (size_t)B_ * TGT_ * V_ + (size_t)2 * B_ * H_;
  for (int s = 0; s < TGT_; ++s) {
    int cur = 1 - pp;
    embed_kernel<<<B_, 256, 0, stream>>>(tok, embed, xdec);
    gru_cell_kernel<<<gruGrid, 192, 0, stream>>>(
        xdec, H_, H_,
        h0b[pp], h0f[pp], Wihb[2], Whhb[2], bih[2], bhh[2],
        h0f[cur], h0b[cur], (__bf16*)nullptr, 0L);
    gru_cell_kernel<<<gruGrid, 192, 0, stream>>>(
        h0b[cur], H_, H_,
        h1b[pp], h1f[pp], Wihb[3], Whhb[3], bih[3], bhh[3],
        h1f[cur], h1b[cur], combb + H_, (long)2 * H_);  // fills comb[:,H:2H]
    // q = h1 @ Wq^T + bq
    gemm_bias_kernel<<<dim3(B_ / 16, H_ / 64), 128, panelH, stream>>>(
        h1b[cur], H_, Wqb, H_, bq, qf, (long)H_, 0L, (__bf16*)nullptr, 0L);
    attn_score_kernel<<<B_, 256, 0, stream>>>(qf, encb, scor);
    softmax_attn_kernel<<<B_, 256, 0, stream>>>(scor, aw, attn_out, s);
    att_sum_kernel<<<B_, 256, 0, stream>>>(aw, encb, combb);  // comb[:,0:H]
    // cvec = comb @ Wc^T + bc  (bf16 out, feeds FC)
    gemm_bias_kernel<<<dim3(B_ / 16, H_ / 64), 128, panel2H, stream>>>(
        combb, 2 * H_, Wcb, 2 * H_, bc,
        (float*)nullptr, 0L, 0L, cvecb, (long)H_);
    // logits -> d_out[b, s, :]
    gemm_bias_kernel<<<dim3(B_ / 16, V_ / 64), 128, panelH, stream>>>(
        cvecb, H_, Wfcb, H_, bfc,
        dout, (long)TGT_ * V_, (long)s * V_, (__bf16*)nullptr, 0L);
    argmax_kernel<<<B_, 256, 0, stream>>>(dout + (size_t)s * V_,
                                          (long)TGT_ * V_, tok);
    pp = cur;
  }

  // hidden = stack([h0, h1]) at offset B*TGT*V
  copy_hidden_kernel<<<(B_ * H_ + 255) / 256, 256, 0, stream>>>(
      h0f[pp], h1f[pp], dout + (size_t)B_ * TGT_ * V_);
}
